// DiffAttention_49847390437777
// MI455X (gfx1250) — matrix operand order repaired and checked by reading, not compiled, not run on hardware
//
#include <hip/hip_runtime.h>
#include <hip/hip_bf16.h>
#include <stdint.h>
#include <stddef.h>
#include <type_traits>

// ---- problem constants ----
#define S_LEN    2048
#define NHEADS   16
#define HDIM     128
#define NPAIRS   8
#define QTILE    64
#define KVBLK    32
#define LAMBDA_INIT 0.77787009955925598f   // 0.8 - 0.6*exp(-3.3)
#define OUT_SCALE   (1.0f - LAMBDA_INIT)
#define SCALE_LOG2  0.12751744561878443f   // (1/sqrt(128)) * log2(e)  (exp2-domain softmax)
#define NEG_BIG    -1.0e30f

#define KB_ELEMS ((size_t)2 * NHEADS * S_LEN * HDIM)   // 8,388,608 bf16
#define KB_BYTES (KB_ELEMS * 2)                        // 16 MB per tensor

typedef __bf16 v8bf  __attribute__((ext_vector_type(8)));
typedef __bf16 v16bf __attribute__((ext_vector_type(16)));
typedef float  v8f   __attribute__((ext_vector_type(8)));

// Double-buffered K/V staging (bf16), overlaid with the f32 epilogue exchange tile.
struct SmemMain {
  __bf16 Ks [2][KVBLK][HDIM];   // 16 KB : K rows, heads 2p+0 / 2p+1
  __bf16 Vts[2][HDIM][KVBLK];   // 16 KB : V^T,  kv-heads 2p+0 / 2p+1
};
struct SmemEpi { float Es[4][16][128]; };               // 32 KB
union SmemBig { SmemMain m[2]; SmemEpi e; };            // 64 KB
static_assert(sizeof(SmemMain) == 32768, "smem layout");

// 16-lane XOR butterflies via ds_swizzle (group-of-32 encoding: and=0x1f, xor=mask<<10)
#define XSWZ(x, imm) __int_as_float(__builtin_amdgcn_ds_swizzle(__float_as_int(x), (imm)))
__device__ __forceinline__ float redmax16(float v) {
  v = fmaxf(v, XSWZ(v, 0x041F)); v = fmaxf(v, XSWZ(v, 0x081F));
  v = fmaxf(v, XSWZ(v, 0x101F)); v = fmaxf(v, XSWZ(v, 0x201F));
  return v;
}
__device__ __forceinline__ float redsum16(float v) {
  v += XSWZ(v, 0x041F); v += XSWZ(v, 0x081F);
  v += XSWZ(v, 0x101F); v += XSWZ(v, 0x201F);
  return v;
}

__device__ __forceinline__ v16bf make_frag(const __bf16* p0, const __bf16* p1) {
  union { v16bf v; v8bf h[2]; } u;
  u.h[0] = *reinterpret_cast<const v8bf*>(p0);
  u.h[1] = *reinterpret_cast<const v8bf*>(p1);
  return u.v;
}

// CDNA5 async global->LDS copy, 16B per lane, tracked by ASYNCcnt.
__device__ __forceinline__ void async_cp16(uint32_t lds_byte, const void* gsrc) {
  asm volatile("global_load_async_to_lds_b128 %0, %1, off"
               :: "v"(lds_byte), "v"(gsrc) : "memory");
}

// ---------- pre-pass: one-time f32 -> bf16 conversion into workspace ----------
// kb : [bi][h][s][d]            (row-major per head; contiguous 8KB per 32-row block)
// vtb: [bi][h][blk][d][kv32]    (pre-transposed to WMMA B-operand order; 8KB blocks)
__global__ __launch_bounds__(256)
void convert_kv(const float* __restrict__ k, const float* __restrict__ v,
                __bf16* __restrict__ kb, __bf16* __restrict__ vtb) {
  const size_t idx4 = (size_t)blockIdx.x * 256 + threadIdx.x;   // 2,097,152 total
  const size_t e0 = idx4 * 4;                                   // (bi, s, h, d) flat
  const int d0 = (int)(e0 & 127);
  size_t t = e0 >> 7;
  const int h = (int)(t & 15);
  t >>= 4;
  const int s  = (int)(t & 2047);
  const int bi = (int)(t >> 11);
  const float4 fk = *reinterpret_cast<const float4*>(k + e0);
  const float4 fv = *reinterpret_cast<const float4*>(v + e0);
  __bf16* kd = kb + (((size_t)(bi * NHEADS + h)) * S_LEN + s) * HDIM + d0;
  kd[0] = (__bf16)fk.x; kd[1] = (__bf16)fk.y; kd[2] = (__bf16)fk.z; kd[3] = (__bf16)fk.w;
  __bf16* vd = vtb + ((((size_t)(bi * NHEADS + h)) * (S_LEN / KVBLK) + (s >> 5)) * HDIM + d0)
                       * KVBLK + (s & 31);
  vd[0]  = (__bf16)fv.x; vd[32] = (__bf16)fv.y;
  vd[64] = (__bf16)fv.z; vd[96] = (__bf16)fv.w;
}

// ---------- main fused differential-attention kernel ----------
__global__ __launch_bounds__(256, 1)
void diff_attn_wmma(const float* __restrict__ q,
                    const float* __restrict__ lq1, const float* __restrict__ lk1,
                    const float* __restrict__ lq2, const float* __restrict__ lk2,
                    const float* __restrict__ sw, const unsigned char* __restrict__ amask,
                    const __bf16* __restrict__ kbuf, const __bf16* __restrict__ vtbuf,
                    float* __restrict__ out) {
  __shared__ __align__(16) SmemBig sm;
  __shared__ __align__(16) __bf16 Ps[8][16][KVBLK];   // per-wave P C->A relayout (8 KB)
  __shared__ float lred1[128], lred2[128];
  __shared__ float lamS;
  __shared__ float biasS[2][KVBLK];

  const int tid  = threadIdx.x;
  const int lane = tid & 31;
  const int n    = lane & 15, hi = lane >> 4;
  // wave index as an SGPR -> all wave-level predicates become scalar branches
  const int wave = __builtin_amdgcn_readfirstlane(tid >> 5);
  const int head_sel = wave >> 2;        // 0 -> head 2p (attn_1), 1 -> head 2p+1 (attn_2)
  const int qsub     = wave & 3;
  const int qt = blockIdx.x, ph = blockIdx.y, bi = blockIdx.z;
  const int head  = ph * 2 + head_sel;
  const int qbase = qt * QTILE + qsub * 16;

  // lambda_full
  if (tid < 128) { lred1[tid] = lq1[tid] * lk1[tid]; lred2[tid] = lq2[tid] * lk2[tid]; }
  __syncthreads();
  if (tid == 0) {
    float s1 = 0.f, s2 = 0.f;
#pragma unroll 8
    for (int i = 0; i < 128; ++i) { s1 += lred1[i]; s2 += lred2[i]; }
    lamS = __expf(s1) - __expf(s2) + LAMBDA_INIT;
  }

  // Q fragments (A-operand layout)
  v16bf qf[4];
  {
    const float* qp = q + (((size_t)(bi * S_LEN + qbase + n)) * NHEADS + head) * HDIM;
#pragma unroll
    for (int c = 0; c < 4; ++c) {
      const int b0 = 32 * c + 8 * hi;
      float4 f0 = *reinterpret_cast<const float4*>(qp + b0);
      float4 f1 = *reinterpret_cast<const float4*>(qp + b0 + 4);
      float4 f2 = *reinterpret_cast<const float4*>(qp + b0 + 16);
      float4 f3 = *reinterpret_cast<const float4*>(qp + b0 + 20);
      float tf[16] = {f0.x,f0.y,f0.z,f0.w, f1.x,f1.y,f1.z,f1.w,
                      f2.x,f2.y,f2.z,f2.w, f3.x,f3.y,f3.z,f3.w};
      v16bf a;
#pragma unroll
      for (int i = 0; i < 16; ++i) a[i] = (__bf16)tf[i];
      qf[c] = a;
    }
  }

  v8f acc[2][8];
  const v8f vzero = {0.f,0.f,0.f,0.f,0.f,0.f,0.f,0.f};
#pragma unroll
  for (int vi = 0; vi < 2; ++vi)
#pragma unroll
    for (int c2 = 0; c2 < 8; ++c2) acc[vi][c2] = vzero;
  float mrun[8], lrun[8];
#pragma unroll
  for (int r = 0; r < 8; ++r) { mrun[r] = NEG_BIG; lrun[r] = 0.f; }

  // async staging of one 32-row kv block into buffer b (4 contiguous 8KB regions)
  auto issue_stage = [&](int kvb, int b) {
    const int blk = kvb >> 5;
    const __bf16* srcs[4];
    srcs[0] = kbuf + (((size_t)(bi * NHEADS + ph * 2 + 0)) * S_LEN + kvb) * HDIM;
    srcs[1] = kbuf + (((size_t)(bi * NHEADS + ph * 2 + 1)) * S_LEN + kvb) * HDIM;
    srcs[2] = vtbuf + (((size_t)(bi * NHEADS + ph * 2 + 0)) * (S_LEN / KVBLK) + blk) * (HDIM * KVBLK);
    srcs[3] = vtbuf + (((size_t)(bi * NHEADS + ph * 2 + 1)) * (S_LEN / KVBLK) + blk) * (HDIM * KVBLK);
    const uint32_t ldsbase = (uint32_t)(uintptr_t)(&sm.m[b]);
#pragma unroll
    for (int i = 0; i < 8; ++i) {                 // 2048 x 16B = 32 KB, 8 per thread
      const int t   = i * 256 + tid;
      const int rgn = t >> 9;
      const int off = (t & 511) << 4;
      async_cp16(ldsbase + rgn * 8192 + off, (const char*)srcs[rgn] + off);
    }
    if (tid < KVBLK)
      biasS[b][tid] = amask[(size_t)bi * S_LEN + kvb + tid] ? 0.0f : NEG_BIG;
  };

  __syncthreads();                 // lamS visible
  const int nb = (qt * QTILE + QTILE) >> 5;   // causal kv extent in 32-blocks
  issue_stage(0, 0);

  for (int j = 0; j < nb; ++j) {
    const int kvb = j * KVBLK;
    const int cur = j & 1;
    if (j + 1 < nb) {
      issue_stage(kvb + KVBLK, cur ^ 1);       // prefetch next block into other buffer
      asm volatile("s_wait_asynccnt 0x8" ::: "memory");   // current 8 copies done (in-order)
    } else {
      asm volatile("s_wait_asynccnt 0x0" ::: "memory");
    }
    __syncthreads();

    if (kvb <= qbase + 15) {                   // scalar branch (qbase is SGPR)
      // ---- S = Q*K^T : 8 WMMAs ----
      v8f sblk[2];
#pragma unroll
      for (int t = 0; t < 2; ++t) {
        v8f c = vzero;
#pragma unroll
        for (int cc = 0; cc < 4; ++cc) {
          const __bf16* kp = &sm.m[cur].Ks[head_sel][t * 16 + n][cc * 32 + hi * 8];
          v16bf kb = make_frag(kp, kp + 16);
          c = __builtin_amdgcn_wmma_f32_16x16x32_bf16(false, qf[cc], false, kb,
                                                      (short)0, c, false, false);
        }
        sblk[t] = c;
      }
      // ---- online softmax (exp2 domain); alpha collected as a vector ----
      const float b0 = biasS[cur][n], b1 = biasS[cur][16 + n];
      const int kv0 = kvb + n, kv1 = kvb + 16 + n, qrow0 = qbase + 8 * hi;
      v8f alphav;
      auto smx = [&](auto cz) {
        constexpr bool CAUSAL = decltype(cz)::value;
#pragma unroll
        for (int r = 0; r < 8; ++r) {
          float s0 = sblk[0][r] * SCALE_LOG2 + b0;
          float s1 = sblk[1][r] * SCALE_LOG2 + b1;
          if constexpr (CAUSAL) {
            const int qr = qrow0 + r;
            s0 += (kv0 > qr) ? NEG_BIG : 0.f;
            s1 += (kv1 > qr) ? NEG_BIG : 0.f;
          }
          float rm = redmax16(fmaxf(s0, s1));
          float mn = fmaxf(mrun[r], rm);
          float alpha = __builtin_amdgcn_exp2f(mrun[r] - mn);
          mrun[r] = mn;
          alphav[r] = alpha;
          float e0 = __builtin_amdgcn_exp2f(s0 - mn);
          float e1 = __builtin_amdgcn_exp2f(s1 - mn);
          lrun[r] = lrun[r] * alpha + redsum16(e0 + e1);
          Ps[wave][r + 8 * hi][n]      = (__bf16)e0;
          Ps[wave][r + 8 * hi][16 + n] = (__bf16)e1;
        }
      };
      if (kvb + KVBLK - 1 > qbase) smx(std::true_type{});   // diagonal block(s) only
      else                         smx(std::false_type{});
      // vectorized accumulator rescale (packed f32 muls instead of 128 scalars)
#pragma unroll
      for (int vi = 0; vi < 2; ++vi)
#pragma unroll
        for (int c2 = 0; c2 < 8; ++c2) acc[vi][c2] *= alphav;
      // ---- P back in A-operand layout ----
      const __bf16* pp = &Ps[wave][n][0];
      v16bf pf = make_frag(pp + 16 * hi, pp + 16 * hi + 8);
      // ---- PV: 16 WMMAs ----
#pragma unroll
      for (int vi = 0; vi < 2; ++vi)
#pragma unroll
        for (int c2 = 0; c2 < 8; ++c2) {
          const __bf16* vp = &sm.m[cur].Vts[vi][c2 * 16 + n][hi * 16];
          v16bf vb = make_frag(vp, vp + 8);
          acc[vi][c2] = __builtin_amdgcn_wmma_f32_16x16x32_bf16(false, pf, false, vb,
                                                               (short)0, acc[vi][c2],
                                                               false, false);
        }
    }
    __syncthreads();
  }

  // ---- normalize (vectorized) ----
  v8f rinvv;
#pragma unroll
  for (int r = 0; r < 8; ++r) rinvv[r] = 1.0f / lrun[r];
#pragma unroll
  for (int vi = 0; vi < 2; ++vi)
#pragma unroll
    for (int c2 = 0; c2 < 8; ++c2) acc[vi][c2] *= rinvv;

  // ---- epilogue: pair-difference, RMS over 256 dims, store ----
  const float lam = lamS;
  float ssq[8] = {0.f,0.f,0.f,0.f,0.f,0.f,0.f,0.f};
#pragma unroll
  for (int vi = 0; vi < 2; ++vi) {
    if (head_sel == 1) {
#pragma unroll
      for (int c2 = 0; c2 < 8; ++c2)
#pragma unroll
        for (int r = 0; r < 8; ++r)
          sm.e.Es[qsub][r + 8 * hi][c2 * 16 + n] = acc[vi][c2][r];
    }
    __syncthreads();
    if (head_sel == 0) {
#pragma unroll
      for (int c2 = 0; c2 < 8; ++c2)
#pragma unroll
        for (int r = 0; r < 8; ++r) {
          float e  = sm.e.Es[qsub][r + 8 * hi][c2 * 16 + n];
          float dv = acc[vi][c2][r] - lam * e;
          acc[vi][c2][r] = dv;
          ssq[r] += dv * dv;
        }
    }
    __syncthreads();
  }

  if (head_sel == 0) {
    float wgt[2][8];
#pragma unroll
    for (int vi = 0; vi < 2; ++vi)
#pragma unroll
      for (int c2 = 0; c2 < 8; ++c2) wgt[vi][c2] = sw[vi * 128 + c2 * 16 + n];
#pragma unroll
    for (int r = 0; r < 8; ++r) {
      float tot = redsum16(ssq[r]);
      float rms = rsqrtf(tot * (1.0f / 256.0f) + 1e-5f);
      const size_t orow = (((size_t)(bi * S_LEN + qbase + r + 8 * hi)) * NPAIRS + ph) * 256;
#pragma unroll
      for (int vi = 0; vi < 2; ++vi)
#pragma unroll
        for (int c2 = 0; c2 < 8; ++c2)
          out[orow + vi * 128 + c2 * 16 + n] = acc[vi][c2][r] * rms * wgt[vi][c2] * OUT_SCALE;
    }
  }
}

extern "C" void kernel_launch(void* const* d_in, const int* in_sizes, int n_in,
                              void* d_out, int out_size, void* d_ws, size_t ws_size,
                              hipStream_t stream) {
  (void)in_sizes; (void)n_in; (void)out_size; (void)ws_size;
  const float* q   = (const float*)d_in[0];
  const float* k   = (const float*)d_in[1];
  const float* v   = (const float*)d_in[2];
  const float* lq1 = (const float*)d_in[3];
  const float* lk1 = (const float*)d_in[4];
  const float* lq2 = (const float*)d_in[5];
  const float* lk2 = (const float*)d_in[6];
  const float* sw  = (const float*)d_in[7];
  const unsigned char* amask = (const unsigned char*)d_in[8];
  float* out = (float*)d_out;

  __bf16* kbuf  = (__bf16*)d_ws;                       // 16 MB
  __bf16* vtbuf = (__bf16*)((char*)d_ws + KB_BYTES);   // 16 MB

  convert_kv<<<dim3((unsigned)(KB_ELEMS / 4 / 256)), dim3(256), 0, stream>>>(k, v, kbuf, vtbuf);
  diff_attn_wmma<<<dim3(S_LEN / QTILE, NPAIRS, 2), dim3(256), 0, stream>>>(
      q, lq1, lk1, lq2, lk2, sw, amask, kbuf, vtbuf, out);
}
